// MultiHeadSelfAttention_22385369547184
// MI455X (gfx1250) — matrix operand order repaired and checked by reading, not compiled, not run on hardware
//
#include <hip/hip_runtime.h>

// ---------------------------------------------------------------------------
// Types / WMMA helpers (CDNA5 gfx1250, wave32)
// ---------------------------------------------------------------------------
typedef __bf16 bf16;
typedef __bf16 v16bf __attribute__((ext_vector_type(16)));
typedef float  v8f   __attribute__((ext_vector_type(8)));

union Frag { v16bf v; uint4 q[2]; };

__device__ __forceinline__ v8f wmma_bf16(v16bf a, v16bf b, v8f c) {
  // D = A(16x32 bf16) * B(32x16 bf16) + C(16x16 f32)
  return __builtin_amdgcn_wmma_f32_16x16x32_bf16(false, a, false, b,
                                                 (short)0, c, false, false);
}

__device__ __forceinline__ unsigned short f2bf_bits(float f) {
  unsigned u = __float_as_uint(f);
  u += 0x7fffu + ((u >> 16) & 1u);          // round-to-nearest-even
  return (unsigned short)(u >> 16);
}

__device__ __forceinline__ bf16 f2bf(float f) {
  unsigned short s = f2bf_bits(f);
  bf16 r;
  __builtin_memcpy(&r, &s, 2);
  return r;
}

// ---------------------------------------------------------------------------
// Problem constants
// ---------------------------------------------------------------------------
#define BATCH 4
#define SEQ   2048
#define DMODEL 1024
#define NHEAD 16
#define DHEAD 64
#define MTOT  (BATCH * SEQ)      // 8192

// ---------------------------------------------------------------------------
// Kernel 0: fp32 -> bf16 conversion, 4 elements/thread
// ---------------------------------------------------------------------------
__global__ void cvt_f32_to_bf16_v4(const float4* __restrict__ src,
                                   ushort4* __restrict__ dst, int n4) {
  int i = blockIdx.x * blockDim.x + threadIdx.x;
  if (i < n4) {
    float4 f = src[i];
    ushort4 o;
    o.x = f2bf_bits(f.x);
    o.y = f2bf_bits(f.y);
    o.z = f2bf_bits(f.z);
    o.w = f2bf_bits(f.w);
    dst[i] = o;
  }
}

// ---------------------------------------------------------------------------
// Kernel 1/3: GEMM  Y[m,n] = sum_k A[m,k] * B[n,k]   (i.e. Y = A @ B^T)
//   A: [MTOT x 1024] bf16 row-major, B: [1024 x 1024] bf16 row-major.
//   SPLIT=true : write bf16 to head-split [B,H,T,64] (QKV projections)
//   SPLIT=false: write fp32 row-major [MTOT,1024]   (output projection)
// Block tile 128x128, K-step 64. 4 waves (128 threads), wave tile 64x64:
// 16 WMMA accumulators per wave, 1.0 ds_load_b128 per WMMA.
// ---------------------------------------------------------------------------
#define TAS 72   // padded LDS stride (elements)

template<bool SPLIT>
__global__ __launch_bounds__(128) void gemm_bf16_k(
    const bf16* __restrict__ A, const bf16* __restrict__ B,
    bf16* __restrict__ outS, float* __restrict__ outF) {
  __shared__ bf16 As[128 * TAS];
  __shared__ bf16 Bs[128 * TAS];

  const int tid  = threadIdx.x;
  const int wave = tid >> 5;
  const int lane = tid & 31;
  const int half = lane >> 4;
  const int ln   = lane & 15;
  const int wm = (wave & 1) * 64;   // wave M offset inside block tile
  const int wn = (wave >> 1) * 64;  // wave N offset inside block tile
  const int m0 = blockIdx.y * 128;
  const int n0 = blockIdx.x * 128;

  v8f acc[4][4] = {};

  for (int k0 = 0; k0 < DMODEL; k0 += 64) {
    // cooperative tile load: 128 rows x 64 cols, 16B chunks, 8 per thread
#pragma unroll
    for (int i = 0; i < 8; ++i) {
      int c = tid + i * 128;                 // 0..1023
      int row = c >> 3, col = (c & 7) * 8;
      *(uint4*)&As[row * TAS + col] =
          *(const uint4*)&A[(size_t)(m0 + row) * DMODEL + k0 + col];
      *(uint4*)&Bs[row * TAS + col] =
          *(const uint4*)&B[(size_t)(n0 + row) * DMODEL + k0 + col];
    }
    if (k0 + 64 < DMODEL) {   // prefetch next K tile -> global_prefetch_b8
      int c = tid * 8;
      int row = c >> 3, col = (c & 7) * 8;
      __builtin_prefetch(&A[(size_t)(m0 + row) * DMODEL + k0 + 64 + col], 0, 1);
      __builtin_prefetch(&B[(size_t)(n0 + row) * DMODEL + k0 + 64 + col], 0, 1);
    }
    __syncthreads();

#pragma unroll
    for (int kk = 0; kk < 64; kk += 32) {
      Frag af[4];
#pragma unroll
      for (int mt = 0; mt < 4; ++mt) {
        // A frag (16x32): lane m=ln, K chunks at kk+half*8 and kk+16+half*8
        int base = (wm + mt * 16 + ln) * TAS + kk + half * 8;
        af[mt].q[0] = *(const uint4*)&As[base];
        af[mt].q[1] = *(const uint4*)&As[base + 16];
      }
      Frag bfg[4];
#pragma unroll
      for (int nt = 0; nt < 4; ++nt) {
        // B frag (32x16): lane n=ln, K chunks at kk+half*8 and kk+16+half*8
        int base = (wn + nt * 16 + ln) * TAS + kk + half * 8;
        bfg[nt].q[0] = *(const uint4*)&Bs[base];
        bfg[nt].q[1] = *(const uint4*)&Bs[base + 16];
      }
#pragma unroll
      for (int mt = 0; mt < 4; ++mt)
#pragma unroll
        for (int nt = 0; nt < 4; ++nt)
          acc[mt][nt] = wmma_bf16(af[mt].v, bfg[nt].v, acc[mt][nt]);
    }
    __syncthreads();
  }

  // epilogue: C layout -> row = r + 8*half, col = ln
#pragma unroll
  for (int mt = 0; mt < 4; ++mt) {
#pragma unroll
    for (int nt = 0; nt < 4; ++nt) {
#pragma unroll
      for (int r = 0; r < 8; ++r) {
        int row = m0 + wm + mt * 16 + r + half * 8;
        int col = n0 + wn + nt * 16 + ln;
        float v = acc[mt][nt][r];
        if (SPLIT) {
          int b = row >> 11, t = row & (SEQ - 1);
          int h = col >> 6, dh = col & 63;
          outS[((((size_t)b * NHEAD + h) * SEQ + t) << 6) + dh] = f2bf(v);
        } else {
          outF[(size_t)row * DMODEL + col] = v;
        }
      }
    }
  }
}

// ---------------------------------------------------------------------------
// Kernel 2: flash attention.
//   Q,K,V: bf16 [B,H,T,64]; O: bf16 [B,T,1024] (heads re-merged).
//   Block = 128 threads (4 waves); block handles (b,h) x 64 query rows;
//   each wave owns 16 query rows. Streams keys in tiles of 32.
// ---------------------------------------------------------------------------
#define KS 72    // K-tile LDS stride
#define VS 40    // V^T  LDS stride
#define PS 40    // P    LDS stride

__global__ __launch_bounds__(128) void attn_kernel(
    const bf16* __restrict__ Q, const bf16* __restrict__ Kb,
    const bf16* __restrict__ Vb, bf16* __restrict__ O) {
  __shared__ bf16 Kt[32 * KS];          // [key][dh]
  __shared__ bf16 Vt[64 * VS];          // [dh][key]  (transposed)
  __shared__ bf16 Pt[4 * 16 * PS];      // per-wave P staging [row][key]

  const int bh    = blockIdx.y;          // b*16 + h
  const int qbase = blockIdx.x * 64;
  const int tid  = threadIdx.x;
  const int wave = tid >> 5;
  const int lane = tid & 31;
  const int half = lane >> 4;
  const int ln   = lane & 15;
  const int lh8  = half * 8;

  // --- Q fragments pinned in registers (A-matrix 16x32, two K-chunks) ---
  Frag qf[2];
  {
    int qr = qbase + wave * 16 + ln;     // A-matrix row (same for both halves)
    size_t base = ((size_t)bh * SEQ + qr) * DHEAD;
#pragma unroll
    for (int kk = 0; kk < 2; ++kk) {
      qf[kk].q[0] = *(const uint4*)&Q[base + kk * 32 + lh8];
      qf[kk].q[1] = *(const uint4*)&Q[base + kk * 32 + lh8 + 16];
    }
  }

  float run_m[8], run_l[8];
  v8f oacc[4] = {};
#pragma unroll
  for (int r = 0; r < 8; ++r) { run_m[r] = -1e30f; run_l[r] = 0.0f; }

  for (int kt = 0; kt < SEQ; kt += 32) {
    // --- stage K (row-major) and V (transposed) into LDS ---
#pragma unroll
    for (int i = 0; i < 2; ++i) {
      int c = tid + i * 128;             // 0..255 chunks of 8
      int key = c >> 3, col = (c & 7) * 8;
      *(uint4*)&Kt[key * KS + col] =
          *(const uint4*)&Kb[((size_t)bh * SEQ + kt + key) * DHEAD + col];
    }
#pragma unroll
    for (int i = 0; i < 2; ++i) {
      int c = tid + i * 128;
      int key = c >> 3, col = (c & 7) * 8;
      uint4 raw = *(const uint4*)&Vb[((size_t)bh * SEQ + kt + key) * DHEAD + col];
      const bf16* pv = (const bf16*)&raw;
#pragma unroll
      for (int j = 0; j < 8; ++j) Vt[(col + j) * VS + key] = pv[j];
    }
    __syncthreads();

    // --- S = (Q K^T) * 1/sqrt(Dh), 16 rows x 32 keys per wave ---
    v8f sacc[2] = {};
#pragma unroll
    for (int nt = 0; nt < 2; ++nt) {
#pragma unroll
      for (int kk = 0; kk < 2; ++kk) {
        Frag kf;
        int base = (nt * 16 + ln) * KS + kk * 32 + half * 8;
        kf.q[0] = *(const uint4*)&Kt[base];
        kf.q[1] = *(const uint4*)&Kt[base + 16];
        sacc[nt] = wmma_bf16(qf[kk].v, kf.v, sacc[nt]);
      }
    }

    // --- online softmax (row stats live per lane-half) ---
    float p0[8], p1[8], alpha[8];
#pragma unroll
    for (int r = 0; r < 8; ++r) {
      float s0 = sacc[0][r] * 0.125f;    // 1/sqrt(64)
      float s1 = sacc[1][r] * 0.125f;
      float mx = fmaxf(s0, s1);
#pragma unroll
      for (int d = 1; d < 16; d <<= 1) mx = fmaxf(mx, __shfl_xor(mx, d, 32));
      float nm = fmaxf(run_m[r], mx);
      alpha[r] = __expf(run_m[r] - nm);
      p0[r] = __expf(s0 - nm);
      p1[r] = __expf(s1 - nm);
      float rs = p0[r] + p1[r];
#pragma unroll
      for (int d = 1; d < 16; d <<= 1) rs += __shfl_xor(rs, d, 32);
      run_l[r] = run_l[r] * alpha[r] + rs;
      run_m[r] = nm;
    }
#pragma unroll
    for (int dt = 0; dt < 4; ++dt)
#pragma unroll
      for (int r = 0; r < 8; ++r) oacc[dt][r] *= alpha[r];

    // --- P: C-layout -> A-fragment layout via per-wave LDS staging ---
    bf16* P = &Pt[wave * 16 * PS];
#pragma unroll
    for (int r = 0; r < 8; ++r) {
      int row = r + half * 8;
      P[row * PS + ln]      = f2bf(p0[r]);
      P[row * PS + 16 + ln] = f2bf(p1[r]);
    }
    Frag pf;   // DS ops are in-order within a wave; same-wave RAW is safe
    {
      int base = ln * PS + lh8;
      pf.q[0] = *(const uint4*)&P[base];
      pf.q[1] = *(const uint4*)&P[base + 16];
    }

    // --- O += P @ V  (B frag = column dh of V = row of Vt) ---
#pragma unroll
    for (int dt = 0; dt < 4; ++dt) {
      Frag vf;
      int base = (dt * 16 + ln) * VS + half * 8;
      vf.q[0] = *(const uint4*)&Vt[base];
      vf.q[1] = *(const uint4*)&Vt[base + 16];
      oacc[dt] = wmma_bf16(pf.v, vf.v, oacc[dt]);
    }
    __syncthreads();
  }

  // --- normalize and write O back merged: [B,T, h*64+dh] ---
  const int b = bh >> 4, h = bh & 15;
#pragma unroll
  for (int r = 0; r < 8; ++r) {
    float inv = 1.0f / run_l[r];
    int row = qbase + wave * 16 + r + half * 8;
    size_t obase = ((size_t)(b * SEQ + row)) * DMODEL + h * DHEAD;
#pragma unroll
    for (int dt = 0; dt < 4; ++dt)
      O[obase + dt * 16 + ln] = f2bf(oacc[dt][r] * inv);
  }
}

// ---------------------------------------------------------------------------
// Launcher
// ---------------------------------------------------------------------------
extern "C" void kernel_launch(void* const* d_in, const int* in_sizes, int n_in,
                              void* d_out, int out_size, void* d_ws, size_t ws_size,
                              hipStream_t stream) {
  (void)in_sizes; (void)n_in; (void)out_size; (void)ws_size;
  const float* x  = (const float*)d_in[0];
  const float* Wq = (const float*)d_in[1];
  const float* Wk = (const float*)d_in[2];
  const float* Wv = (const float*)d_in[3];
  const float* Wo = (const float*)d_in[4];

  const size_t MD = (size_t)MTOT * DMODEL;   // 8192*1024
  const size_t WD = (size_t)DMODEL * DMODEL; // 1024*1024

  bf16* xb  = (bf16*)d_ws;
  bf16* wqb = xb  + MD;
  bf16* wkb = wqb + WD;
  bf16* wvb = wkb + WD;
  bf16* wob = wvb + WD;
  bf16* Qb  = wob + WD;
  bf16* Kb  = Qb  + MD;
  bf16* Vb  = Kb  + MD;
  bf16* Ob  = Vb  + MD;     // total ~88 MiB of workspace

  // 1) fp32 -> bf16 (4-wide)
  {
    int n4 = (int)(MD / 4);
    cvt_f32_to_bf16_v4<<<(n4 + 255) / 256, 256, 0, stream>>>(
        (const float4*)x, (ushort4*)xb, n4);
    int nw4 = (int)(WD / 4);
    cvt_f32_to_bf16_v4<<<(nw4 + 255) / 256, 256, 0, stream>>>(
        (const float4*)Wq, (ushort4*)wqb, nw4);
    cvt_f32_to_bf16_v4<<<(nw4 + 255) / 256, 256, 0, stream>>>(
        (const float4*)Wk, (ushort4*)wkb, nw4);
    cvt_f32_to_bf16_v4<<<(nw4 + 255) / 256, 256, 0, stream>>>(
        (const float4*)Wv, (ushort4*)wvb, nw4);
    cvt_f32_to_bf16_v4<<<(nw4 + 255) / 256, 256, 0, stream>>>(
        (const float4*)Wo, (ushort4*)wob, nw4);
  }

  // 2) QKV projections (write head-split bf16)
  dim3 gg(DMODEL / 128, MTOT / 128);   // (8, 64)
  gemm_bf16_k<true><<<gg, 128, 0, stream>>>(xb, wqb, Qb, nullptr);
  gemm_bf16_k<true><<<gg, 128, 0, stream>>>(xb, wkb, Kb, nullptr);
  gemm_bf16_k<true><<<gg, 128, 0, stream>>>(xb, wvb, Vb, nullptr);

  // 3) flash attention
  attn_kernel<<<dim3(SEQ / 64, BATCH * NHEAD), 128, 0, stream>>>(Qb, Kb, Vb, Ob);

  // 4) output projection (fp32 out)
  gemm_bf16_k<false><<<gg, 128, 0, stream>>>(Ob, wob, nullptr, (float*)d_out);
}
